// PraxisAttention_51127290692266
// MI455X (gfx1250) — matrix operand order repaired and checked by reading, not compiled, not run on hardware
//
#include <hip/hip_runtime.h>
#include <hip/hip_bf16.h>

typedef _Float16 h8   __attribute__((ext_vector_type(8)));
typedef _Float16 v16h __attribute__((ext_vector_type(16)));
typedef float    v8f  __attribute__((ext_vector_type(8)));

#define BB 2
#define SS 2048
#define DD 1024
#define HH 16
#define DH 64
#define NEGV (-1e9f)

// ---------------- WMMA helpers ----------------

__device__ __forceinline__ v8f wmma_f16(v16h a, v16h b, v8f c) {
  return __builtin_amdgcn_wmma_f32_16x16x32_f16(false, a, false, b, (short)0, c,
                                                false, false);
}

// A-matrix fragment (16xK=32, f16), from row-major [rows][ld] memory.
// lane L: row = row0 + (L&15); halves 0..7 at k0+(L>>4)*8, halves 8..15 at +16.
__device__ __forceinline__ v16h load_a(const _Float16* __restrict__ base, int ld,
                                       int row0, int k0) {
  int lane = threadIdx.x & 31;
  const _Float16* p =
      base + (size_t)(row0 + (lane & 15)) * ld + k0 + ((lane >> 4) << 3);
  h8 lo = *(const h8*)(p);
  h8 hi = *(const h8*)(p + 16);
  return __builtin_shufflevector(lo, hi, 0, 1, 2, 3, 4, 5, 6, 7, 8, 9, 10, 11,
                                 12, 13, 14, 15);
}

// B-matrix fragment (K=32 x 16, f16) from row-major [n][ld] memory (i.e. B^T).
// lane L: n = n0 + (L&15); halves 0..7 at k0+(L>>4)*8, halves 8..15 at +16.
__device__ __forceinline__ v16h load_b(const _Float16* __restrict__ base, int ld,
                                       int n0, int k0) {
  int lane = threadIdx.x & 31;
  const _Float16* p =
      base + (size_t)(n0 + (lane & 15)) * ld + k0 + ((lane >> 4) << 3);
  h8 lo = *(const h8*)(p);
  h8 hi = *(const h8*)(p + 16);
  return __builtin_shufflevector(lo, hi, 0, 1, 2, 3, 4, 5, 6, 7, 8, 9, 10, 11,
                                 12, 13, 14, 15);
}

__device__ __forceinline__ float xor16(float v) { return __shfl_xor(v, 16, 32); }

// ---------------- prep kernels ----------------

__global__ void k_cvt_f16(const float* __restrict__ in, _Float16* __restrict__ out,
                          int n) {
  int i = blockIdx.x * blockDim.x + threadIdx.x;
  if (i < n) out[i] = (_Float16)in[i];
}

__global__ void k_prep_pos(const long long* __restrict__ tok,
                           const float* __restrict__ amask,
                           float* __restrict__ posf, float* __restrict__ maskadd,
                           int n) {
  int i = blockIdx.x * blockDim.x + threadIdx.x;
  if (i < n) {
    posf[i] = (float)tok[i];
    maskadd[i] = (1.0f - amask[i]) * NEGV;
  }
}

__global__ void k_prep_gn(const float* __restrict__ g, const float* __restrict__ b,
                          float* __restrict__ gsc, float* __restrict__ bbt, int n) {
  int i = blockIdx.x * blockDim.x + threadIdx.x;
  if (i < n) {
    gsc[i] = g[i] * 0.2f;  // * (1 - LAMBDA_INIT)
    bbt[i] = b[i] * 0.2f;
  }
}

__global__ void k_lambda(const float* __restrict__ lq1, const float* __restrict__ lq2,
                         const float* __restrict__ lk1, const float* __restrict__ lk2,
                         float* __restrict__ lam) {
  __shared__ float a[64], c[64];
  int t = threadIdx.x;
  a[t] = lq1[t] * lk1[t];
  c[t] = lq2[t] * lk2[t];
  __syncthreads();
  if (t == 0) {
    float d1 = 0.f, d2 = 0.f;
    for (int i = 0; i < 64; i++) { d1 += a[i]; d2 += c[i]; }
    lam[0] = __expf(d1) - __expf(d2) + 0.8f;
  }
}

// ---------------- projection GEMMs ----------------
// C = A(4096,1024) @ W(N,1024)^T, A/W f16 row-major.
// MODE 0: write f16 to [B,H,S,128] (Q/K, N=2048).  MODE 1: write f16 to Vt [B,H,Dh,S].
template <int MODE>
__global__ void __launch_bounds__(256) k_gemm_proj(const _Float16* __restrict__ A,
                                                   const _Float16* __restrict__ W,
                                                   _Float16* __restrict__ out, int N,
                                                   float oscale) {
  const int K = DD, M = BB * SS;
  int wave = blockIdx.x * (blockDim.x >> 5) + (threadIdx.x >> 5);
  int mtiles = M / 32, ntiles = N / 64;
  if (wave >= mtiles * ntiles) return;
  int mt = wave % mtiles, nt = wave / mtiles;
  int m0 = mt * 32, n0 = nt * 64;

  v8f acc[2][4] = {};
  for (int k0 = 0; k0 < K; k0 += 32) {
    v16h a0 = load_a(A, K, m0, k0);
    v16h a1 = load_a(A, K, m0 + 16, k0);
#pragma unroll
    for (int j = 0; j < 4; j++) {
      v16h bf = load_b(W, K, n0 + j * 16, k0);
      acc[0][j] = wmma_f16(a0, bf, acc[0][j]);
      acc[1][j] = wmma_f16(a1, bf, acc[1][j]);
    }
  }
  int lane = threadIdx.x & 31;
  int ln = lane & 15, lh = (lane >> 4) << 3;
#pragma unroll
  for (int i = 0; i < 2; i++)
#pragma unroll
    for (int j = 0; j < 4; j++)
#pragma unroll
      for (int r = 0; r < 8; r++) {
        int m = m0 + i * 16 + lh + r;
        int n = n0 + j * 16 + ln;
        float v = acc[i][j][r] * oscale;
        int b = m >> 11, s = m & (SS - 1);
        if (MODE == 0) {
          int h = n >> 7, c = n & 127;
          out[(((size_t)(b * HH + h) * SS + s) << 7) + c] = (_Float16)v;
        } else {
          int h = n >> 6, d = n & 63;
          out[(((size_t)(b * HH + h) * DH + d) << 11) + s] = (_Float16)v;
        }
      }
}

// ---------------- attention ----------------
// One wave per (b,h, 16-query tile). Scores computed transposed (S^T = K*Q^T) so
// online softmax stats/rescale are per-lane; output accumulated as out^T = V^T*P^T.

__device__ __forceinline__ void online_update(v8f s0, v8f s1, float& mrun,
                                              float& lrun, v8f acc[4],
                                              const _Float16* __restrict__ VtB,
                                              int kb, bool lowhalf) {
  float tmax = -1e30f;
#pragma unroll
  for (int r = 0; r < 8; r++) tmax = fmaxf(tmax, fmaxf(s0[r], s1[r]));
  tmax = fmaxf(tmax, xor16(tmax));
  float mn = fmaxf(mrun, tmax);
  float alpha = __expf(mrun - mn);
  float lsum = 0.f;
  v16h p;
#pragma unroll
  for (int r = 0; r < 8; r++) {
    float e0 = __expf(s0[r] - mn);   // tile0: keys kb + (lane/16)*8 + r
    float e1 = __expf(s1[r] - mn);   // tile1: keys kb+16 + (lane/16)*8 + r
    lsum += e0 + e1;
    float e0p = xor16(e0);
    float e1p = xor16(e1);
    // B-frag layout: elems 0..7 keys (lane/16)*8+r, elems 8..15 keys 16+(lane/16)*8+r
    float lo = e0;
    float hi = e1;
    p[r] = (_Float16)lo;
    p[r + 8] = (_Float16)hi;
  }
  lsum += xor16(lsum);
  lrun = lrun * alpha + lsum;
  mrun = mn;
#pragma unroll
  for (int dt = 0; dt < 4; dt++) {
    acc[dt] = acc[dt] * alpha;
    v16h av = load_a(VtB, SS, dt * 16, kb);
    acc[dt] = wmma_f16(av, p, acc[dt]);
  }
}

__global__ void __launch_bounds__(256) k_attn(const _Float16* __restrict__ Qh,
                                              const _Float16* __restrict__ Kh,
                                              const _Float16* __restrict__ Vt,
                                              const float* __restrict__ posf,
                                              const float* __restrict__ maskadd,
                                              const float* __restrict__ lamp,
                                              float* __restrict__ attn) {
  int wave = blockIdx.x * 8 + (threadIdx.x >> 5);
  int qt = wave & 127;
  int bh = wave >> 7;
  int b = bh >> 4, h = bh & 15;
  int lane = threadIdx.x & 31, ln = lane & 15;
  bool lowhalf = lane < 16;
  int lh8 = (lane >> 4) << 3;

  const _Float16* Qb_ = Qh + (size_t)bh * SS * 128;
  const _Float16* Kb_ = Kh + (size_t)bh * SS * 128;
  const _Float16* Vb_ = Vt + (size_t)bh * DH * SS;

  int q0 = qt * 16;
  int q = q0 + ln;
  float posq = posf[b * SS + q];
  float slope = exp2f(-0.5f * (float)(h + 1));  // 2^(-8*(h+1)/H)

  v16h Qf[4];
#pragma unroll
  for (int c = 0; c < 4; c++) Qf[c] = load_b(Qb_, 128, q0, c * 32);

  float m1 = -1e30f, l1 = 0.f, m2 = -1e30f, l2 = 0.f;
  v8f acc1[4] = {}, acc2[4] = {};
  v8f vz = {};

  int kend = q0 + 15;
  for (int kb = 0; kb <= kend; kb += 32) {
    v8f s1t[2], s2t[2];
#pragma unroll
    for (int t = 0; t < 2; t++) {
      int kt = kb + t * 16;
      v8f c0 = wmma_f16(load_a(Kb_, 128, kt, 0), Qf[0], vz);
      s1t[t] = wmma_f16(load_a(Kb_, 128, kt, 32), Qf[1], c0);
      v8f c2 = wmma_f16(load_a(Kb_, 128, kt, 64), Qf[2], vz);
      s2t[t] = wmma_f16(load_a(Kb_, 128, kt, 96), Qf[3], c2);
#pragma unroll
      for (int r = 0; r < 8; r++) {
        int key = kt + lh8 + r;
        float av = slope * (posf[b * SS + key] - posq) + maskadd[b * SS + key] +
                   (key > q ? NEGV : 0.f);
        s1t[t][r] += av;
        s2t[t][r] += av;
      }
    }
    online_update(s1t[0], s1t[1], m1, l1, acc1, Vb_, kb, lowhalf);
    online_update(s2t[0], s2t[1], m2, l2, acc2, Vb_, kb, lowhalf);
  }

  float lam = lamp[0];
  float i1 = 1.f / l1, i2 = 1.f / l2;
  float* op = attn + ((size_t)bh * SS + q) * DH;
#pragma unroll
  for (int dt = 0; dt < 4; dt++)
#pragma unroll
    for (int r = 0; r < 8; r++) {
      int d = dt * 16 + lh8 + r;
      op[d] = acc1[dt][r] * i1 - lam * acc2[dt][r] * i2;
    }
}

// ---------------- group-norm stats (deterministic tree reduction) -------------

__global__ void __launch_bounds__(256) k_gn_stats(const float* __restrict__ attn,
                                                  float* __restrict__ meanA,
                                                  float* __restrict__ rstdA) {
  int bh = blockIdx.x;
  const float* p = attn + (size_t)bh * SS * DH;
  double s = 0.0, s2 = 0.0;
  for (int i = threadIdx.x; i < SS * DH; i += 256) {
    float v = p[i];
    s += v;
    s2 += (double)v * (double)v;
  }
  __shared__ double sh[256], sh2[256];
  sh[threadIdx.x] = s;
  sh2[threadIdx.x] = s2;
  __syncthreads();
  for (int st = 128; st > 0; st >>= 1) {
    if (threadIdx.x < st) {
      sh[threadIdx.x] += sh[threadIdx.x + st];
      sh2[threadIdx.x] += sh2[threadIdx.x + st];
    }
    __syncthreads();
  }
  if (threadIdx.x == 0) {
    double inv = 1.0 / (double)(SS * DH);
    double mean = sh[0] * inv;
    double var = sh2[0] * inv - mean * mean;
    meanA[bh] = (float)mean;
    rstdA[bh] = (float)(1.0 / sqrt(var + 1e-5));
  }
}

// ---------------- final GEMM: out = norm(attn) @ Wo^T ----------------

__global__ void __launch_bounds__(256) k_gemm_out(
    const float* __restrict__ attn, const _Float16* __restrict__ Wo,
    const float* __restrict__ gsc, const float* __restrict__ bbt,
    const float* __restrict__ meanA, const float* __restrict__ rstdA,
    float* __restrict__ out) {
  const int K = DD, M = BB * SS, N = DD;
  int wave = blockIdx.x * (blockDim.x >> 5) + (threadIdx.x >> 5);
  int mtiles = M / 32, ntiles = N / 64;
  if (wave >= mtiles * ntiles) return;
  int mt = wave % mtiles, nt = wave / mtiles;
  int m0 = mt * 32, n0 = nt * 64;
  int lane = threadIdx.x & 31, ln = lane & 15, lup = lane >> 4;
  int bidx = m0 >> 11;  // batch uniform per wave (32 | 2048)

  v8f acc[2][4] = {};
  for (int k0 = 0; k0 < K; k0 += 32) {
    int hh = k0 >> 6;  // head group uniform for this 32-chunk
    float mean = meanA[bidx * HH + hh];
    float rstd = rstdA[bidx * HH + hh];
    v16h a[2];
#pragma unroll
    for (int i = 0; i < 2; i++) {
      int m = m0 + i * 16 + ln;
      int s = m & (SS - 1);
      int co = (k0 & 63) + lup * 8;
      const float* ap =
          attn + (((size_t)(bidx * HH + hh) * SS + s) << 6) + co;
      int j0 = k0 + lup * 8;
#pragma unroll
      for (int t = 0; t < 8; t++) {
        float xv = ap[t];
        float yv = ap[16 + t];
        a[i][t] = (_Float16)((xv - mean) * rstd * gsc[j0 + t] + bbt[j0 + t]);
        a[i][t + 8] =
            (_Float16)((yv - mean) * rstd * gsc[j0 + 16 + t] + bbt[j0 + 16 + t]);
      }
    }
#pragma unroll
    for (int j = 0; j < 4; j++) {
      v16h bf = load_b(Wo, K, n0 + j * 16, k0);
      acc[0][j] = wmma_f16(a[0], bf, acc[0][j]);
      acc[1][j] = wmma_f16(a[1], bf, acc[1][j]);
    }
  }
  int lh = lup << 3;
#pragma unroll
  for (int i = 0; i < 2; i++)
#pragma unroll
    for (int j = 0; j < 4; j++)
#pragma unroll
      for (int r = 0; r < 8; r++) {
        int m = m0 + i * 16 + lh + r;
        int n = n0 + j * 16 + ln;
        out[(size_t)m * DD + n] = acc[i][j][r];
      }
}

// ---------------- launcher ----------------

extern "C" void kernel_launch(void* const* d_in, const int* in_sizes, int n_in,
                              void* d_out, int out_size, void* d_ws, size_t ws_size,
                              hipStream_t stream) {
  const float* X = (const float*)d_in[0];
  const float* amask = (const float*)d_in[1];
  const long long* tok = (const long long*)d_in[2];
  const float* Wq = (const float*)d_in[3];
  const float* Wk = (const float*)d_in[4];
  const float* Wv = (const float*)d_in[5];
  const float* Wo = (const float*)d_in[6];
  const float* lq1 = (const float*)d_in[7];
  const float* lq2 = (const float*)d_in[8];
  const float* lk1 = (const float*)d_in[9];
  const float* lk2 = (const float*)d_in[10];
  const float* gng = (const float*)d_in[11];
  const float* gnb = (const float*)d_in[12];
  float* outp = (float*)d_out;

  size_t off = 0;
  char* base = (char*)d_ws;
  auto alloc = [&](size_t bytes) -> void* {
    void* p = base + off;
    off += (bytes + 255) & ~(size_t)255;
    return p;
  };
  _Float16* Xh = (_Float16*)alloc((size_t)BB * SS * DD * 2);
  _Float16* Wqh = (_Float16*)alloc((size_t)2 * HH * DH * DD * 2);
  _Float16* Wkh = (_Float16*)alloc((size_t)2 * HH * DH * DD * 2);
  _Float16* Wvh = (_Float16*)alloc((size_t)HH * DH * DD * 2);
  _Float16* Woh = (_Float16*)alloc((size_t)DD * HH * DH * 2);
  _Float16* Qh = (_Float16*)alloc((size_t)BB * HH * SS * 128 * 2);
  _Float16* Kh = (_Float16*)alloc((size_t)BB * HH * SS * 128 * 2);
  _Float16* Vt = (_Float16*)alloc((size_t)BB * HH * DH * SS * 2);
  float* attnW = (float*)alloc((size_t)BB * HH * SS * DH * 4);
  float* posf = (float*)alloc((size_t)BB * SS * 4);
  float* maskadd = (float*)alloc((size_t)BB * SS * 4);
  float* gsc = (float*)alloc(DD * 4);
  float* bbt = (float*)alloc(DD * 4);
  float* meanA = (float*)alloc(BB * HH * 4);
  float* rstdA = (float*)alloc(BB * HH * 4);
  float* lamp = (float*)alloc(4);

  // --- prep ---
  {
    int n = BB * SS * DD;
    k_cvt_f16<<<(n + 255) / 256, 256, 0, stream>>>(X, Xh, n);
  }
  {
    int n = 2 * HH * DH * DD;
    k_cvt_f16<<<(n + 255) / 256, 256, 0, stream>>>(Wq, Wqh, n);
    k_cvt_f16<<<(n + 255) / 256, 256, 0, stream>>>(Wk, Wkh, n);
  }
  {
    int n = HH * DH * DD;
    k_cvt_f16<<<(n + 255) / 256, 256, 0, stream>>>(Wv, Wvh, n);
    k_cvt_f16<<<(n + 255) / 256, 256, 0, stream>>>(Wo, Woh, n);
  }
  k_prep_pos<<<(BB * SS + 255) / 256, 256, 0, stream>>>(tok, amask, posf, maskadd,
                                                        BB * SS);
  k_prep_gn<<<(DD + 255) / 256, 256, 0, stream>>>(gng, gnb, gsc, bbt, DD);
  k_lambda<<<1, 64, 0, stream>>>(lq1, lq2, lk1, lk2, lamp);

  // --- projections (scale 1/sqrt(Dh) folded into Q) ---
  k_gemm_proj<0><<<512, 256, 0, stream>>>(Xh, Wqh, Qh, 2048, 0.125f);
  k_gemm_proj<0><<<512, 256, 0, stream>>>(Xh, Wkh, Kh, 2048, 1.0f);
  k_gemm_proj<1><<<256, 256, 0, stream>>>(Xh, Wvh, Vt, 1024, 1.0f);

  // --- differential flash attention ---
  k_attn<<<512, 256, 0, stream>>>(Qh, Kh, Vt, posf, maskadd, lamp, attnW);

  // --- group-norm stats + final projection ---
  k_gn_stats<<<BB * HH, 256, 0, stream>>>(attnW, meanA, rstdA);
  k_gemm_out<<<256, 256, 0, stream>>>(attnW, Woh, gsc, bbt, meanA, rstdA, outp);

  (void)in_sizes; (void)n_in; (void)out_size; (void)ws_size;
}
